// Decoder_Model_89369679495586
// MI455X (gfx1250) — matrix operand order repaired and checked by reading, not compiled, not run on hardware
//
#include <hip/hip_runtime.h>
#include <hip/hip_bf16.h>

typedef __bf16 bf16;
typedef __attribute__((ext_vector_type(16))) __bf16 v16bf;
typedef __attribute__((ext_vector_type(8)))  __bf16 v8bf;
typedef __attribute__((ext_vector_type(4)))  __bf16 v4bf;
typedef __attribute__((ext_vector_type(8)))  float  v8f;

__device__ __forceinline__ v8f vzero8() {
    v8f z = {0.f, 0.f, 0.f, 0.f, 0.f, 0.f, 0.f, 0.f};
    return z;
}

__device__ __forceinline__ v8f wmma_bf16(v16bf a, v16bf b, v8f c) {
    return __builtin_amdgcn_wmma_f32_16x16x32_bf16(
        /*neg_a=*/false, a, /*neg_b=*/false, b,
        /*c_mod=*/(short)0, c, /*reuse_a=*/false, /*reuse_b=*/false);
}

// A fragment (16x32 bf16, MxK): lanes 0-15 hold K[kb..kb+7] and K[kb+16..kb+23]
// with kb = (lane/16)*8.  Two 16B LDS loads.
__device__ __forceinline__ v16bf frag_a(const bf16* base, int ld, int row, int k0, int lane) {
    const int mr = lane & 15;
    const int kb = (lane >> 4) << 3;
    const bf16* p = base + (size_t)(row + mr) * ld + k0 + kb;
    v8bf lo = *(const v8bf*)(p);
    v8bf hi = *(const v8bf*)(p + 16);
    v16bf r;
#pragma unroll
    for (int i = 0; i < 8; ++i) { r[i] = lo[i]; r[i + 8] = hi[i]; }
    return r;
}

// B fragment (32x16 bf16, KxN): lane holds column n = lane%16,
// K[kb..kb+7] and K[kb+16..kb+23] with kb = k0 + (lane/16)*8  (two 16B LDS loads).
__device__ __forceinline__ v16bf frag_b(const bf16* base, int ld, int col, int k0, int lane) {
    const int nr = lane & 15;
    const int kh = (lane >> 4) << 3;
    const bf16* p = base + (size_t)(col + nr) * ld + k0 + kh;
    v8bf lo = *(const v8bf*)(p);
    v8bf hi = *(const v8bf*)(p + 16);
    v16bf r;
#pragma unroll
    for (int i = 0; i < 8; ++i) { r[i] = lo[i]; r[i + 8] = hi[i]; }
    return r;
}

// ---------------------------------------------------------------------------
// GEMM: Y[M,N] = X[M,K] @ W[N,K]^T + bias (+ReLU FLAGS&1) (+residual FLAGS&2)
// Block tile 128x128, K-step 32, 256 threads = 8 waves arranged 4x2,
// each wave computes 32x64 via 2x4 WMMA fragments.
// N, K, FLAGS are compile-time: addressing lowers to shifts, no branches.
// Software-pipelined: next K-tile global loads issue before the WMMA stack.
// ---------------------------------------------------------------------------
#define LDT 40  // LDS row stride in halves (80B, 16B aligned, conflict-spread)

template <int N, int K, int FLAGS>
__global__ __launch_bounds__(256) void gemm_bf16w(
    const float* __restrict__ X, const float* __restrict__ W,
    const float* __restrict__ bias, const float* __restrict__ resid,
    float* __restrict__ Y)
{
    __shared__ bf16 Xs[128 * LDT];
    __shared__ bf16 Wsh[128 * LDT];

    const int t = threadIdx.x;
    const int lane = t & 31;
    const int w = t >> 5;
    const int wm = w >> 1;   // 0..3
    const int wn = w & 1;    // 0..1
    const int bm0 = blockIdx.y * 128;
    const int bn0 = blockIdx.x * 128;

    // per-thread staging coordinates (4 float4 per matrix per K-tile)
    int srow[4], scol[4];
#pragma unroll
    for (int r = 0; r < 4; ++r) {
        int idx = t + r * 256;          // 1024 float4s
        srow[r] = idx >> 3;
        scol[r] = (idx & 7) << 2;
    }

    v8f acc[2][4];
#pragma unroll
    for (int mi = 0; mi < 2; ++mi)
#pragma unroll
        for (int ni = 0; ni < 4; ++ni) acc[mi][ni] = vzero8();

    // prefetch K-tile 0 into registers
    float4 px[4], pw[4];
#pragma unroll
    for (int r = 0; r < 4; ++r) {
        px[r] = *(const float4*)(X + (size_t)(bm0 + srow[r]) * K + scol[r]);
        pw[r] = *(const float4*)(W + (size_t)(bn0 + srow[r]) * K + scol[r]);
    }

    for (int k0 = 0; k0 < K; k0 += 32) {
        __syncthreads();   // previous compute's LDS reads complete
        // stash prefetched tile into LDS as bf16
#pragma unroll
        for (int r = 0; r < 4; ++r) {
            v4bf xb = {(bf16)px[r].x, (bf16)px[r].y, (bf16)px[r].z, (bf16)px[r].w};
            *(v4bf*)(Xs + srow[r] * LDT + scol[r]) = xb;
            v4bf wb = {(bf16)pw[r].x, (bf16)pw[r].y, (bf16)pw[r].z, (bf16)pw[r].w};
            *(v4bf*)(Wsh + srow[r] * LDT + scol[r]) = wb;
        }
        __syncthreads();

        // issue next tile's global loads, then overlap them with the WMMAs
        if (k0 + 32 < K) {
            int kn = k0 + 32;
#pragma unroll
            for (int r = 0; r < 4; ++r) {
                px[r] = *(const float4*)(X + (size_t)(bm0 + srow[r]) * K + kn + scol[r]);
                pw[r] = *(const float4*)(W + (size_t)(bn0 + srow[r]) * K + kn + scol[r]);
            }
        }

        v16bf av[2], bv[4];
#pragma unroll
        for (int mi = 0; mi < 2; ++mi) av[mi] = frag_a(Xs, LDT, wm * 32 + mi * 16, 0, lane);
#pragma unroll
        for (int ni = 0; ni < 4; ++ni) bv[ni] = frag_b(Wsh, LDT, wn * 64 + ni * 16, 0, lane);
#pragma unroll
        for (int mi = 0; mi < 2; ++mi)
#pragma unroll
            for (int ni = 0; ni < 4; ++ni)
                acc[mi][ni] = wmma_bf16(av[mi], bv[ni], acc[mi][ni]);
    }

    const int jr = (lane >> 4) << 3;
#pragma unroll
    for (int mi = 0; mi < 2; ++mi) {
#pragma unroll
        for (int ni = 0; ni < 4; ++ni) {
            int col = bn0 + wn * 64 + ni * 16 + (lane & 15);
            float bb = bias[col];
#pragma unroll
            for (int j = 0; j < 8; ++j) {
                int row = bm0 + wm * 32 + mi * 16 + jr + j;
                float y = acc[mi][ni][j] + bb;
                if (FLAGS & 1) y = fmaxf(y, 0.0f);
                if (FLAGS & 2) y += resid[(size_t)row * N + col];
                Y[(size_t)row * N + col] = y;
            }
        }
    }
}

// ---------------------------------------------------------------------------
// Flash attention: per (b, h, 128 q-rows). Q/K/V are (B*S, 512) with head h
// occupying columns [h*64, h*64+64). scores = QK^T / 32 (+ causal mask),
// online softmax, @V. Output written at [b*S+s][h*64+d].
// ---------------------------------------------------------------------------
#define SQ 1024
#define DM 512
#define LQ 72    // K/Q LDS stride (halves)
#define LP 136   // P/Vt LDS stride (halves)

template <int CAUSAL>
__global__ __launch_bounds__(256) void attn_wmma(
    const float* __restrict__ Q, const float* __restrict__ Kv,
    const float* __restrict__ V, float* __restrict__ O)
{
    __shared__ bf16 Qs[128 * LQ];
    __shared__ bf16 Ks[128 * LQ];
    __shared__ bf16 Vt[64 * LP];   // V transposed: [d][kv]
    __shared__ bf16 Ps[128 * LP];  // probabilities, per-wave 16-row slabs

    const int t = threadIdx.x;
    const int lane = t & 31;
    const int w = t >> 5;
    const int nl = lane & 15;
    const int jr = (lane >> 4) << 3;
    const int qblk = blockIdx.x, h = blockIdx.y, b = blockIdx.z;
    const int qb0 = qblk * 128;

    const float* Qp = Q + (size_t)b * SQ * DM + h * 64;
    const float* Kp = Kv + (size_t)b * SQ * DM + h * 64;
    const float* Vp = V + (size_t)b * SQ * DM + h * 64;

    // stage Q block (128x64) into LDS as bf16
#pragma unroll
    for (int r = 0; r < 8; ++r) {
        int idx = t + r * 256;     // 2048 float4s
        int row = idx >> 4;
        int c4  = (idx & 15) << 2;
        float4 xv = *(const float4*)(Qp + (size_t)(qb0 + row) * DM + c4);
        v4bf xb = {(bf16)xv.x, (bf16)xv.y, (bf16)xv.z, (bf16)xv.w};
        *(v4bf*)(Qs + row * LQ + c4) = xb;
    }
    __syncthreads();

    v16bf aq[2];
    aq[0] = frag_a(Qs, LQ, w * 16, 0, lane);
    aq[1] = frag_a(Qs, LQ, w * 16, 32, lane);

    float rmax[8], rsum[8];
    v8f acc[4];
#pragma unroll
    for (int j = 0; j < 8; ++j) { rmax[j] = -3.0e38f; rsum[j] = 0.f; }
#pragma unroll
    for (int dt = 0; dt < 4; ++dt) acc[dt] = vzero8();

    const int nchunks = CAUSAL ? (qblk + 1) : (SQ / 128);
    for (int c = 0; c < nchunks; ++c) {
        const int kv0 = c * 128;
        __syncthreads();   // previous chunk's LDS reads complete
        // stage K (row-major) and V (transposed) chunks
#pragma unroll
        for (int r = 0; r < 8; ++r) {
            int idx = t + r * 256;
            int row = idx >> 4;
            int c4  = (idx & 15) << 2;
            float4 k4 = *(const float4*)(Kp + (size_t)(kv0 + row) * DM + c4);
            v4bf kb4 = {(bf16)k4.x, (bf16)k4.y, (bf16)k4.z, (bf16)k4.w};
            *(v4bf*)(Ks + row * LQ + c4) = kb4;
            float4 v4 = *(const float4*)(Vp + (size_t)(kv0 + row) * DM + c4);
            Vt[(c4 + 0) * LP + row] = (bf16)v4.x;
            Vt[(c4 + 1) * LP + row] = (bf16)v4.y;
            Vt[(c4 + 2) * LP + row] = (bf16)v4.z;
            Vt[(c4 + 3) * LP + row] = (bf16)v4.w;
        }
        __syncthreads();

        // scores: 16 x 128 per wave, 8 kv-tiles x 2 k-steps
        float sc[8][8];
#pragma unroll
        for (int nt = 0; nt < 8; ++nt) {
            v8f s = vzero8();
            v16bf bk0 = frag_b(Ks, LQ, nt * 16, 0, lane);
            v16bf bk1 = frag_b(Ks, LQ, nt * 16, 32, lane);
            s = wmma_bf16(aq[0], bk0, s);
            s = wmma_bf16(aq[1], bk1, s);
            int kvcol = kv0 + nt * 16 + nl;
#pragma unroll
            for (int j = 0; j < 8; ++j) {
                float val = s[j] * 0.03125f;  // / (DK/2) = /32, faithful to reference
                if (CAUSAL && kvcol > (qb0 + w * 16 + jr + j)) val += -1.0e8f;
                sc[nt][j] = val;
            }
        }

        // online softmax per row (row shared by 16 lanes of a half-wave)
#pragma unroll
        for (int j = 0; j < 8; ++j) {
            float mc = sc[0][j];
#pragma unroll
            for (int nt = 1; nt < 8; ++nt) mc = fmaxf(mc, sc[nt][j]);
#pragma unroll
            for (int off = 8; off >= 1; off >>= 1)
                mc = fmaxf(mc, __shfl_xor(mc, off, 16));
            float mnew = fmaxf(rmax[j], mc);
            float scl = __expf(rmax[j] - mnew);
            rmax[j] = mnew;
            float ssum = 0.f;
#pragma unroll
            for (int nt = 0; nt < 8; ++nt) {
                float p = __expf(sc[nt][j] - mnew);
                sc[nt][j] = p;
                ssum += p;
            }
#pragma unroll
            for (int off = 8; off >= 1; off >>= 1)
                ssum += __shfl_xor(ssum, off, 16);
            rsum[j] = rsum[j] * scl + ssum;
#pragma unroll
            for (int dt = 0; dt < 4; ++dt) acc[dt][j] *= scl;
        }

        // re-layout probabilities D->A via LDS (bf16)
#pragma unroll
        for (int nt = 0; nt < 8; ++nt)
#pragma unroll
            for (int j = 0; j < 8; ++j)
                Ps[(w * 16 + jr + j) * LP + nt * 16 + nl] = (bf16)sc[nt][j];
        __syncthreads();

        // acc += P @ V   (4 k-steps of 32 kv, 4 d-tiles)
#pragma unroll
        for (int kk = 0; kk < 4; ++kk) {
            v16bf ap = frag_a(Ps, LP, w * 16, kk * 32, lane);
#pragma unroll
            for (int dt = 0; dt < 4; ++dt) {
                v16bf bvv = frag_b(Vt, LP, dt * 16, kk * 32, lane);
                acc[dt] = wmma_bf16(ap, bvv, acc[dt]);
            }
        }
    }

    float* Op = O + (size_t)b * SQ * DM + h * 64;
#pragma unroll
    for (int dt = 0; dt < 4; ++dt) {
        int col = dt * 16 + nl;
#pragma unroll
        for (int j = 0; j < 8; ++j) {
            int row = qb0 + w * 16 + jr + j;
            Op[(size_t)row * DM + col] = acc[dt][j] / rsum[j];
        }
    }
}

// ---------------------------------------------------------------------------
// Full-tensor layer norm (mean/var over ALL elements), deterministic 2-stage.
// ---------------------------------------------------------------------------
__global__ __launch_bounds__(256) void reduce_partial_k(
    const float* __restrict__ x, float* __restrict__ part, int n)
{
    __shared__ float s1[256], s2[256];
    float a = 0.f, b = 0.f;
    for (size_t i = (size_t)blockIdx.x * 256 + threadIdx.x; i < (size_t)n;
         i += (size_t)gridDim.x * 256) {
        float v = x[i];
        a += v; b += v * v;
    }
    s1[threadIdx.x] = a; s2[threadIdx.x] = b;
    __syncthreads();
    for (int s = 128; s > 0; s >>= 1) {
        if (threadIdx.x < s) {
            s1[threadIdx.x] += s1[threadIdx.x + s];
            s2[threadIdx.x] += s2[threadIdx.x + s];
        }
        __syncthreads();
    }
    if (threadIdx.x == 0) {
        part[blockIdx.x * 2 + 0] = s1[0];
        part[blockIdx.x * 2 + 1] = s2[0];
    }
}

__global__ __launch_bounds__(256) void finalize_k(
    const float* __restrict__ part, float* __restrict__ stats, int nb, float invN)
{
    __shared__ float s1[256], s2[256];
    float a = 0.f, b = 0.f;
    for (int i = threadIdx.x; i < nb; i += 256) {
        a += part[2 * i + 0];
        b += part[2 * i + 1];
    }
    s1[threadIdx.x] = a; s2[threadIdx.x] = b;
    __syncthreads();
    for (int s = 128; s > 0; s >>= 1) {
        if (threadIdx.x < s) {
            s1[threadIdx.x] += s1[threadIdx.x + s];
            s2[threadIdx.x] += s2[threadIdx.x + s];
        }
        __syncthreads();
    }
    if (threadIdx.x == 0) {
        float mu = s1[0] * invN;
        float var = s2[0] * invN - mu * mu;
        stats[0] = mu;
        stats[1] = rsqrtf(var + 1e-5f);
    }
}

__global__ __launch_bounds__(256) void normalize_k(
    const float* __restrict__ x, const float* __restrict__ stats,
    float* __restrict__ y, int n)
{
    float mu = stats[0], is = stats[1];
    for (size_t i = (size_t)blockIdx.x * 256 + threadIdx.x; i < (size_t)n;
         i += (size_t)gridDim.x * 256) {
        y[i] = (x[i] - mu) * is;
    }
}

// ---------------------------------------------------------------------------
extern "C" void kernel_launch(void* const* d_in, const int* in_sizes, int n_in,
                              void* d_out, int out_size, void* d_ws, size_t ws_size,
                              hipStream_t stream)
{
    const float* enc  = (const float*)d_in[0];
    const float* dec  = (const float*)d_in[1];
    const float* wq_m = (const float*)d_in[2];
    const float* bq_m = (const float*)d_in[3];
    const float* wk_m = (const float*)d_in[4];
    const float* bk_m = (const float*)d_in[5];
    const float* wv_m = (const float*)d_in[6];
    const float* bv_m = (const float*)d_in[7];
    const float* wo_m = (const float*)d_in[8];
    const float* bo_m = (const float*)d_in[9];
    const float* wq_c = (const float*)d_in[10];
    const float* bq_c = (const float*)d_in[11];
    const float* wk_c = (const float*)d_in[12];
    const float* bk_c = (const float*)d_in[13];
    const float* wv_c = (const float*)d_in[14];
    const float* bv_c = (const float*)d_in[15];
    const float* wo_c = (const float*)d_in[16];
    const float* bo_c = (const float*)d_in[17];
    const float* wf1  = (const float*)d_in[18];
    const float* bf1  = (const float*)d_in[19];
    const float* wf2  = (const float*)d_in[20];
    const float* bf2  = (const float*)d_in[21];

    const int M = 8192, D = 512, F = 2048;
    const size_t MD = (size_t)M * D;

    float* q     = (float*)d_ws;
    float* kbuf  = q + MD;
    float* vbuf  = kbuf + MD;
    float* attn  = vbuf + MD;
    float* x1    = attn + MD;
    float* n1    = x1 + MD;
    float* n2    = n1 + MD;
    float* hbuf  = n2 + MD;
    float* part  = hbuf + (size_t)M * F;
    float* stats = part + 2048;

    const dim3 gDD(D / 128, M / 128);   // (4, 64)
    const dim3 gDF(F / 128, M / 128);   // (16, 64)

    auto lnorm = [&](const float* x, float* y) {
        int n = (int)MD;
        reduce_partial_k<<<1024, 256, 0, stream>>>(x, part, n);
        finalize_k<<<1, 256, 0, stream>>>(part, stats, 1024, 1.0f / (float)n);
        normalize_k<<<2048, 256, 0, stream>>>(x, stats, y, n);
    };

    // ---- masked self-attention ----
    gemm_bf16w<512, 512, 0><<<gDD, 256, 0, stream>>>(dec, wq_m, bq_m, nullptr, q);
    gemm_bf16w<512, 512, 0><<<gDD, 256, 0, stream>>>(dec, wk_m, bk_m, nullptr, kbuf);
    gemm_bf16w<512, 512, 0><<<gDD, 256, 0, stream>>>(dec, wv_m, bv_m, nullptr, vbuf);
    attn_wmma<1><<<dim3(8, 8, 8), 256, 0, stream>>>(q, kbuf, vbuf, attn);
    gemm_bf16w<512, 512, 2><<<gDD, 256, 0, stream>>>(attn, wo_m, bo_m, dec, x1);  // +resid
    lnorm(x1, n1);                                  // norm_mmhe

    // ---- cross attention ----
    gemm_bf16w<512, 512, 0><<<gDD, 256, 0, stream>>>(n1, wq_c, bq_c, nullptr, q);
    gemm_bf16w<512, 512, 0><<<gDD, 256, 0, stream>>>(enc, wk_c, bk_c, nullptr, kbuf);
    gemm_bf16w<512, 512, 0><<<gDD, 256, 0, stream>>>(enc, wv_c, bv_c, nullptr, vbuf);
    attn_wmma<0><<<dim3(8, 8, 8), 256, 0, stream>>>(q, kbuf, vbuf, attn);
    gemm_bf16w<512, 512, 2><<<gDD, 256, 0, stream>>>(attn, wo_c, bo_c, n1, x1);   // +resid
    lnorm(x1, n2);                                  // norm_mhe

    // ---- FFN ----
    gemm_bf16w<2048, 512, 1><<<gDF, 256, 0, stream>>>(n2, wf1, bf1, nullptr, hbuf);   // ReLU
    gemm_bf16w<512, 2048, 2><<<gDD, 256, 0, stream>>>(hbuf, wf2, bf2, n2, x1);        // +resid
    lnorm(x1, (float*)d_out);

    (void)in_sizes; (void)n_in; (void)out_size; (void)ws_size;
}
